// GatedDeltaRuleFloatCastModel_65017214926851
// MI455X (gfx1250) — matrix operand order repaired and checked by reading, not compiled, not run on hardware
//
#include <hip/hip_runtime.h>

typedef __bf16 bf16_t;
typedef __attribute__((ext_vector_type(8)))  bf16_t bf16x8;
typedef __attribute__((ext_vector_type(16))) bf16_t v16bf;
typedef __attribute__((ext_vector_type(8)))  float  v8f;

#define BATCH 4
#define HEADS 16
#define SEQ   2048
#define DKDIM 128
#define DVDIM 128
#define CHN   16                 // chunk length == WMMA M/N
#define NCH   (SEQ / CHN)        // 128 chunks
#define DVSPLIT 2
#define VROWS (DVDIM / DVSPLIT)  // 64 state rows per block
#define NWAVES (VROWS / 16)      // 4 waves
#define BLOCKT (NWAVES * 32)     // 128 threads
#define DKP   136                // 128 + 8 pad (16B-aligned rows, conflict-free)
#define VCP   72                 // 64 + 8 pad
#define TP    40                 // t-stride: 16 data + 16 zero pad + 8 conflict pad

static __device__ inline v16bf cat8(bf16x8 a, bf16x8 b) {
  return __builtin_shufflevector(a, b, 0,1,2,3,4,5,6,7,8,9,10,11,12,13,14,15);
}

// A-operand (16x32 bf16) from row-major LDS M[row][k]:
//   lane<16 : row=lr, K = kbase+[0..7] , [16..23]
//   lane>=16: row=lr, K = kbase+[8..15], [24..31]
static __device__ inline v16bf load_A16(const bf16_t* base, int stride, int r0,
                                        int kbase, int lr, int hi) {
  const bf16_t* p = base + (size_t)(r0 + lr) * stride + kbase + hi * 8;
  return cat8(*(const bf16x8*)p, *(const bf16x8*)(p + 16));
}

// B-operand (32x16 bf16), B[k][n] = M[n][kbase+k]:
//   lane<16 : col=lr, K = kbase+[0..7] , [16..23]
//   lane>=16: col=lr, K = kbase+[8..15], [24..31]
static __device__ inline v16bf load_B16(const bf16_t* base, int stride, int n0,
                                        int kbase, int lr, int hi) {
  const bf16_t* p = base + (size_t)(n0 + lr) * stride + kbase + hi * 8;
  return cat8(*(const bf16x8*)p, *(const bf16x8*)(p + 16));
}

__global__ __launch_bounds__(BLOCKT)
void gdr_chunked_wmma_kernel(const bf16_t* __restrict__ q, const bf16_t* __restrict__ k,
                             const bf16_t* __restrict__ v, const bf16_t* __restrict__ g,
                             const bf16_t* __restrict__ beta,
                             const float* __restrict__ state0,
                             bf16_t* __restrict__ out) {
  __shared__ bf16_t Kc[CHN][DKP];         // K chunk, row-major [t][d]
  __shared__ bf16_t Qc[CHN][DKP];         // Q chunk
  __shared__ bf16_t Vc[CHN][VCP];         // V chunk, this block's DV slice
  __shared__ bf16_t KT[DKDIM][TP];        // K^T [d][t], cols 16..31 stay zero
  __shared__ bf16_t Sbf[VROWS][DKP];      // bf16 state mirror S[v][d]
  __shared__ bf16_t Ubf[NWAVES][16][TP];  // scaled u [v'][t], cols 16..31 stay zero
  __shared__ float  scr[NWAVES][4][16][17]; // 0:KK^T 1:K.S^T(->u) 2:QK^T 3:Q.S^T
  __shared__ float  gammaS[CHN], rgammaS[CHN], betaS[CHN];

  const int tid  = threadIdx.x;
  const int w    = tid >> 5;
  const int lane = tid & 31;
  const int lr   = lane & 15;
  const int hi   = lane >> 4;
  const int bh     = blockIdx.x >> 1;
  const int vblock = (blockIdx.x & 1) * VROWS;
  const float scale = 0.08838834764831845f; // 128^-0.5

  // ---- one-time: zero the t>=16 pad columns of KT and Ubf ----
  {
    bf16x8 z8;
#pragma unroll
    for (int i = 0; i < 8; ++i) z8[i] = (bf16_t)0.0f;
    *(bf16x8*)&KT[tid][16] = z8;
    *(bf16x8*)&KT[tid][24] = z8;
    if (tid < NWAVES * 16) {
      *(bf16x8*)&Ubf[tid >> 4][tid & 15][16] = z8;
      *(bf16x8*)&Ubf[tid >> 4][tid & 15][24] = z8;
    }
  }

  // ---- init: fp32 state (TRANSPOSED: C rows = d, cols = v') + bf16 mirror ----
  v8f S[8];
#pragma unroll
  for (int tile = 0; tile < 8; ++tile) {
#pragma unroll
    for (int r = 0; r < 8; ++r) {
      int d = tile * 16 + r + 8 * hi;        // C-layout row (M) = d within tile
      int vloc = w * 16 + lr;                // C-layout col (N) = v'
      float s0 = state0[(((size_t)bh * DVDIM) + vblock + vloc) * DKDIM + d];
      S[tile][r]   = s0;
      Sbf[vloc][d] = (bf16_t)s0;
    }
  }

  for (int c = 0; c < NCH; ++c) {
    const int tbase = c * CHN;

    // ---- phase A: cooperative global -> LDS (K, Q, V, K^T, gamma/beta) ----
    {
      int tr = tid >> 3;
      int d0 = (tid & 7) << 4;
      size_t go = ((size_t)bh * SEQ + tbase + tr) * DKDIM + d0;
      bf16x8 k0 = *(const bf16x8*)(k + go);
      bf16x8 k1 = *(const bf16x8*)(k + go + 8);
      *(bf16x8*)&Kc[tr][d0]     = k0;
      *(bf16x8*)&Kc[tr][d0 + 8] = k1;
#pragma unroll
      for (int i = 0; i < 8; ++i) { KT[d0 + i][tr] = k0[i]; KT[d0 + 8 + i][tr] = k1[i]; }
      bf16x8 q0 = *(const bf16x8*)(q + go);
      bf16x8 q1 = *(const bf16x8*)(q + go + 8);
      *(bf16x8*)&Qc[tr][d0]     = q0;
      *(bf16x8*)&Qc[tr][d0 + 8] = q1;
      int v0 = (tid & 7) << 3;
      size_t gv = ((size_t)bh * SEQ + tbase + tr) * DVDIM + vblock + v0;
      *(bf16x8*)&Vc[tr][v0] = *(const bf16x8*)(v + gv);
      if (c + 1 < NCH) {  // pull next chunk toward the caches while we compute
        __builtin_prefetch(k + go + (size_t)CHN * DKDIM, 0, 1);
        __builtin_prefetch(q + go + (size_t)CHN * DKDIM, 0, 1);
        __builtin_prefetch(v + gv + (size_t)CHN * DVDIM, 0, 1);
      }
      if (tid < CHN) {  // cumulative log-decay -> gamma, 1/gamma
        float s = 0.f;
        for (int i = 0; i <= tid; ++i) s += (float)g[(size_t)bh * SEQ + tbase + i];
        gammaS[tid]  = __expf(s);
        rgammaS[tid] = __expf(-s);
        betaS[tid]   = (float)beta[(size_t)bh * SEQ + tbase + tid];
      }
    }
    __syncthreads();

    // ---- phase B: 16 WMMAs per wave: K.S^T, KK^T, QK^T, Q.S^T over DK=128 ----
    v8f m0, agr, wqk, oin;
#pragma unroll
    for (int i = 0; i < 8; ++i) { m0[i] = 0.f; agr[i] = 0.f; wqk[i] = 0.f; oin[i] = 0.f; }
#pragma unroll
    for (int j = 0; j < 4; ++j) {
      int db = j * 32;
      v16bf KA = load_A16(&Kc[0][0],  DKP, 0,      db, lr, hi);
      v16bf QA = load_A16(&Qc[0][0],  DKP, 0,      db, lr, hi);
      v16bf KB = load_B16(&Kc[0][0],  DKP, 0,      db, lr, hi);
      v16bf SB = load_B16(&Sbf[0][0], DKP, w * 16, db, lr, hi);
      m0  = __builtin_amdgcn_wmma_f32_16x16x32_bf16(false, KA, false, SB, (short)0, m0,  false, false);
      agr = __builtin_amdgcn_wmma_f32_16x16x32_bf16(false, KA, false, KB, (short)0, agr, false, false);
      wqk = __builtin_amdgcn_wmma_f32_16x16x32_bf16(false, QA, false, KB, (short)0, wqk, false, false);
      oin = __builtin_amdgcn_wmma_f32_16x16x32_bf16(false, QA, false, SB, (short)0, oin, false, false);
    }
#pragma unroll
    for (int r = 0; r < 8; ++r) {  // C-layout (rows=t, cols=v'/s) -> scratch
      int row = r + 8 * hi;
      scr[w][0][row][lr] = agr[r];
      scr[w][1][row][lr] = m0[r];
      scr[w][2][row][lr] = wqk[r];
      scr[w][3][row][lr] = oin[r];
    }
    __syncthreads();

    // ---- phase C1: forward substitution u=(I+A)^-1 b (lanes 0..15) ----
    if (lane < 16) {
      const int vp = lane;
      float u[CHN];
      float gl = gammaS[CHN - 1];
#pragma unroll
      for (int t = 0; t < CHN; ++t) {
        float bt  = betaS[t];
        float acc = bt * ((float)Vc[t][w * 16 + vp] - gammaS[t] * scr[w][1][t][vp]);
#pragma unroll
        for (int s = 0; s < t; ++s)
          acc -= bt * gammaS[t] * rgammaS[s] * scr[w][0][t][s] * u[s];
        u[t] = acc;
        scr[w][1][t][vp] = acc;              // broadcast u for phase C2
      }
      bf16x8 p0, p1;                         // scaled u: (gamma_L/gamma_t)*u_t
#pragma unroll
      for (int t = 0; t < 8; ++t) {
        p0[t] = (bf16_t)(gl * rgammaS[t] * u[t]);
        p1[t] = (bf16_t)(gl * rgammaS[t + 8] * u[t + 8]);
      }
      *(bf16x8*)&Ubf[w][vp][0] = p0;
      *(bf16x8*)&Ubf[w][vp][8] = p1;
    }
    __syncthreads();

    // ---- phase C2: output o_t (all 32 lanes; lane halves take interleaved t) ----
    {
      const int vp = lr;
#pragma unroll
      for (int tt = 0; tt < 8; ++tt) {
        int t = 2 * tt + hi;
        float gt = gammaS[t];
        float o  = gt * scr[w][3][t][vp];
#pragma unroll
        for (int s = 0; s < CHN; ++s) {
          float cf = (s <= t) ? gt * rgammaS[s] * scr[w][2][t][s] : 0.f;
          o += cf * scr[w][1][s][vp];
        }
        out[((size_t)bh * SEQ + tbase + t) * DVDIM + vblock + w * 16 + vp] =
            (bf16_t)(o * scale);
      }
    }

    // ---- phase D: state update S^T = gamma_L*S^T + K^T @ Utilde (8 WMMAs) ----
    {
      float gl = gammaS[CHN - 1];
      v16bf UB = load_B16(&Ubf[w][0][0], TP, 0, 0, lr, hi);  // t>=16 half is LDS zeros
#pragma unroll
      for (int tile = 0; tile < 8; ++tile) {
#pragma unroll
        for (int r = 0; r < 8; ++r) S[tile][r] *= gl;
        v16bf KA2 = load_A16(&KT[0][0], TP, tile * 16, 0, lr, hi);  // t>=16 zeros
        S[tile] = __builtin_amdgcn_wmma_f32_16x16x32_bf16(false, KA2, false, UB,
                                                          (short)0, S[tile], false, false);
        bf16x8 pk;                         // contiguous d's of row v' -> one b128 store
#pragma unroll
        for (int r = 0; r < 8; ++r) pk[r] = (bf16_t)S[tile][r];
        *(bf16x8*)&Sbf[w * 16 + lr][tile * 16 + 8 * hi] = pk;
      }
    }
    __syncthreads();
  }
}

extern "C" void kernel_launch(void* const* d_in, const int* in_sizes, int n_in,
                              void* d_out, int out_size, void* d_ws, size_t ws_size,
                              hipStream_t stream) {
  (void)in_sizes; (void)n_in; (void)out_size; (void)d_ws; (void)ws_size;
  const bf16_t* q    = (const bf16_t*)d_in[0];
  const bf16_t* kk   = (const bf16_t*)d_in[1];
  const bf16_t* v    = (const bf16_t*)d_in[2];
  const bf16_t* g    = (const bf16_t*)d_in[3];
  const bf16_t* beta = (const bf16_t*)d_in[4];
  const float*  st   = (const float*)d_in[5];
  bf16_t* out = (bf16_t*)d_out;
  dim3 grid(BATCH * HEADS * DVSPLIT);
  dim3 block(BLOCKT);
  gdr_chunked_wmma_kernel<<<grid, block, 0, stream>>>(q, kk, v, g, beta, st, out);
}
